// PointNetSetUpConv_33724083208691
// MI455X (gfx1250) — matrix operand order repaired and checked by reading, not compiled, not run on hardware
//
#include <hip/hip_runtime.h>

typedef unsigned short ushort_t;
typedef __attribute__((ext_vector_type(16))) __bf16 v16bf;
typedef __attribute__((ext_vector_type(8)))  float  v8f;
typedef __attribute__((ext_vector_type(4)))  int    v4i;

#define EPS_BN 1e-5f

#ifndef __has_builtin
#define __has_builtin(x) 0
#endif

// Native f32 -> bf16 (RNE) conversion; backend picks v_cvt_*_bf16_f32.
__device__ __forceinline__ ushort_t f2bf(float f) {
  return __builtin_bit_cast(ushort_t, (__bf16)f);
}

// Async Global->LDS copy of 16 bytes per lane (GLOBAL_LOAD_ASYNC_TO_LDS_B128,
// tracked by ASYNCcnt). Falls back to a VGPR round-trip if builtin is absent.
__device__ __forceinline__ void async_copy_b128(const void* gsrc, void* ldst) {
#if __has_builtin(__builtin_amdgcn_global_load_async_to_lds_b128)
  __builtin_amdgcn_global_load_async_to_lds_b128(
      (__attribute__((address_space(1))) v4i*)(unsigned long long)gsrc,
      (__attribute__((address_space(3))) v4i*)ldst, 0, 0);
#else
  *(uint4*)ldst = *(const uint4*)gsrc;
#endif
}

__device__ __forceinline__ void wait_async_lds() {
#if __has_builtin(__builtin_amdgcn_global_load_async_to_lds_b128)
#if __has_builtin(__builtin_amdgcn_s_wait_asynccnt)
  __builtin_amdgcn_s_wait_asynccnt(0);
#else
  __asm__ volatile("s_wait_asynccnt 0" ::: "memory");
#endif
#endif
  __asm__ volatile("s_wait_dscnt 0" ::: "memory");
}

union Frag { uint4 u[2]; v16bf v; };

// A fragment (16-bit, 16x32): lane<16 -> row M=lane, K {k0..k0+7, k0+16..k0+23};
// lane>=16 -> same M, K halves shifted by 8. A stored row-major [m][k] in LDS.
__device__ __forceinline__ v16bf load_a_frag(const ushort_t* a, int m, int kpad,
                                             int k0, int sel) {
  Frag f;
  f.u[0] = *(const uint4*)(a + m * kpad + k0 + sel * 8);
  f.u[1] = *(const uint4*)(a + m * kpad + k0 + 16 + sel * 8);
  return f.v;
}

// B fragment: lane<16 -> col N, K {k0..k0+7, k0+16..k0+23}; lane>=16 -> K halves shifted by 8.
__device__ __forceinline__ v16bf load_b_frag(const ushort_t* wrow, int k0, int sel) {
  Frag f;
  f.u[0] = *(const uint4*)(wrow + k0 + sel * 8);
  f.u[1] = *(const uint4*)(wrow + k0 + 16 + sel * 8);
  return f.v;
}

__device__ __forceinline__ v8f wmma_bf16(v16bf a, v16bf b, v8f c) {
  return __builtin_amdgcn_wmma_f32_16x16x32_bf16(false, a, false, b, (short)0, c,
                                                 false, false);
}

template<int KS>
__device__ __forceinline__ v8f gemm_tile(const ushort_t* aw, int akpad,
                                         const ushort_t* wrow, int lane) {
  int m = lane & 15, sel = lane >> 4;
  v8f c = {0.f, 0.f, 0.f, 0.f, 0.f, 0.f, 0.f, 0.f};
#pragma unroll
  for (int ks = 0; ks < KS; ++ks) {
    v16bf a = load_a_frag(aw, m, akpad, ks * 32, sel);
    v16bf b = load_b_frag(wrow, ks * 32, sel);
    c = wmma_bf16(a, b, c);
  }
  return c;
}

// ---------------- kNN: one thread per query, xyz2 staged in LDS -------------
__global__ __launch_bounds__(256) void knn_kernel(const float* __restrict__ xyz1,
                                                  const float* __restrict__ xyz2,
                                                  int* __restrict__ idx) {
  __shared__ float s2[2048 * 3];
  int b = blockIdx.y;
  for (int i = threadIdx.x; i < 2048 * 3; i += 256)
    s2[i] = xyz2[(size_t)b * 2048 * 3 + i];
  __syncthreads();
  int q = blockIdx.x * 256 + threadIdx.x;
  const float* p = xyz1 + ((size_t)b * 8192 + q) * 3;
  float px = p[0], py = p[1], pz = p[2];
  float bd[8]; int bi[8];
#pragma unroll
  for (int t = 0; t < 8; ++t) { bd[t] = 3.4e38f; bi[t] = 0; }
  for (int j = 0; j < 2048; ++j) {
    float dx = s2[j*3+0] - px, dy = s2[j*3+1] - py, dz = s2[j*3+2] - pz;
    float d = dx*dx + dy*dy + dz*dz;
    if (d < bd[7]) {
      bd[7] = d; bi[7] = j;
#pragma unroll
      for (int t = 7; t > 0; --t)
        if (bd[t] < bd[t-1]) {
          float td = bd[t]; bd[t] = bd[t-1]; bd[t-1] = td;
          int   ti = bi[t]; bi[t] = bi[t-1]; bi[t-1] = ti;
        }
    }
  }
#pragma unroll
  for (int t = 0; t < 8; ++t) idx[((size_t)b * 8192 + q) * 8 + t] = bi[t];
}

// ------------- (B,C,N) f32 -> (B,N,C) bf16 so gathers are contiguous --------
__global__ __launch_bounds__(256) void transpose_cast_kernel(
    const float* __restrict__ src, ushort_t* __restrict__ dst, int C, int N) {
  int b = blockIdx.z, c = blockIdx.y;
  int n = blockIdx.x * 256 + threadIdx.x;
  dst[((size_t)b * N + n) * C + c] = f2bf(src[((size_t)b * C + c) * N + n]);
}

// --------- fused gather + MLP1 (3 layers, BN+ReLU) + max over k=8 -----------
// Block: 256 thr (8 waves), 32 queries. Wave: 2 subtiles of 16 rows (2q x 8s).
__global__ __launch_bounds__(256) void mlp1_kernel(
    const float* __restrict__ xyz1, const float* __restrict__ xyz2,
    const int* __restrict__ idx, const ushort_t* __restrict__ f2T,
    const float* __restrict__ w0, const float* __restrict__ w1,
    const float* __restrict__ w2,
    const float* __restrict__ bn0, const float* __restrict__ bn1,
    const float* __restrict__ bn2, ushort_t* __restrict__ pooled) {
  extern __shared__ char smem[];
  ushort_t* wB0 = (ushort_t*)smem;            // 128 x 288 (K padded 259->288)
  ushort_t* wB1 = wB0 + 128 * 288;            // 128 x 128
  ushort_t* wB2 = wB1 + 128 * 128;            // 256 x 128
  float2* sbn0 = (float2*)(wB2 + 256 * 128);  // folded BN: (scale, shift)
  float2* sbn1 = sbn0 + 128;
  float2* sbn2 = sbn1 + 128;
  ushort_t* aT = (ushort_t*)(sbn2 + 256);     // 8 waves x 16 x 288

  int tid = threadIdx.x, lane = tid & 31, wave = tid >> 5;
  int b = blockIdx.y, qblk = blockIdx.x * 32;

  for (int i = tid; i < 128 * 288; i += 256) {
    int n = i / 288, k = i - n * 288;
    wB0[i] = (k < 259) ? f2bf(w0[n * 259 + k]) : (ushort_t)0;
  }
  for (int i = tid; i < 128 * 128; i += 256) wB1[i] = f2bf(w1[i]);
  for (int i = tid; i < 256 * 128; i += 256) wB2[i] = f2bf(w2[i]);
  if (tid < 128) {
    float s = bn0[tid] * rsqrtf(bn0[384 + tid] + EPS_BN);
    sbn0[tid] = make_float2(s, bn0[128 + tid] - bn0[256 + tid] * s);
    s = bn1[tid] * rsqrtf(bn1[384 + tid] + EPS_BN);
    sbn1[tid] = make_float2(s, bn1[128 + tid] - bn1[256 + tid] * s);
  }
  if (tid < 256) {
    float s = bn2[tid] * rsqrtf(bn2[768 + tid] + EPS_BN);
    sbn2[tid] = make_float2(s, bn2[256 + tid] - bn2[512 + tid] * s);
  }
  __syncthreads();

  ushort_t* aw = aT + wave * 16 * 288;

  for (int sub = 0; sub < 2; ++sub) {
    int q0 = qblk + wave * 4 + sub * 2;
    {   // gather A tile: 16 rows x (256 feat + 3 posdiff + pad); 2 lanes/row
      int row = lane >> 1, half = lane & 1;
      int q = q0 + (row >> 3), s = row & 7;
      int j = idx[((size_t)b * 8192 + q) * 8 + s];
      const ushort_t* src = f2T + ((size_t)b * 2048 + j) * 256 + half * 128;
      ushort_t* dst = aw + row * 288 + half * 128;
#pragma unroll
      for (int t = 0; t < 16; ++t)          // 16 x 16B async gather into LDS
        async_copy_b128(src + t * 8, dst + t * 8);
      if (half) {
        const float* p2 = xyz2 + ((size_t)b * 2048 + j) * 3;
        const float* p1 = xyz1 + ((size_t)b * 8192 + q) * 3;
        uint4 z0;
        z0.x = (unsigned)f2bf(p2[0] - p1[0]) | ((unsigned)f2bf(p2[1] - p1[1]) << 16);
        z0.y = (unsigned)f2bf(p2[2] - p1[2]); z0.z = 0u; z0.w = 0u;
        uint4 zz = {0u, 0u, 0u, 0u};
        uint4* pd = (uint4*)(aw + row * 288 + 256);
        pd[0] = z0; pd[1] = zz; pd[2] = zz; pd[3] = zz;
      }
    }
    wait_async_lds();

    v8f acc[8];
    int mbase = (lane >> 4) * 8;
    // layer 1: 259(->288) -> 128
#pragma unroll
    for (int nt = 0; nt < 8; ++nt)
      acc[nt] = gemm_tile<9>(aw, 288, wB0 + (nt * 16 + (lane & 15)) * 288, lane);
#pragma unroll
    for (int nt = 0; nt < 8; ++nt) {
      int c = nt * 16 + (lane & 15);
      float2 st = sbn0[c];
#pragma unroll
      for (int r = 0; r < 8; ++r)
        aw[(mbase + r) * 288 + c] = f2bf(fmaxf(acc[nt][r] * st.x + st.y, 0.f));
    }
    __asm__ volatile("s_wait_dscnt 0" ::: "memory");
    // layer 2: 128 -> 128
#pragma unroll
    for (int nt = 0; nt < 8; ++nt)
      acc[nt] = gemm_tile<4>(aw, 288, wB1 + (nt * 16 + (lane & 15)) * 128, lane);
#pragma unroll
    for (int nt = 0; nt < 8; ++nt) {
      int c = nt * 16 + (lane & 15);
      float2 st = sbn1[c];
#pragma unroll
      for (int r = 0; r < 8; ++r)
        aw[(mbase + r) * 288 + c] = f2bf(fmaxf(acc[nt][r] * st.x + st.y, 0.f));
    }
    __asm__ volatile("s_wait_dscnt 0" ::: "memory");
    // layer 3: 128 -> 256, BN+ReLU, then max over 8 samples (the 8 acc VGPRs)
#pragma unroll
    for (int h = 0; h < 2; ++h) {
#pragma unroll
      for (int nt = 0; nt < 8; ++nt)
        acc[nt] = gemm_tile<4>(aw, 288,
                               wB2 + ((h * 8 + nt) * 16 + (lane & 15)) * 128, lane);
      int q = q0 + (lane >> 4);
#pragma unroll
      for (int nt = 0; nt < 8; ++nt) {
        int c = (h * 8 + nt) * 16 + (lane & 15);
        float2 st = sbn2[c];
        float m = 0.f;  // ReLU output is >= 0
#pragma unroll
        for (int r = 0; r < 8; ++r)
          m = fmaxf(m, fmaxf(acc[nt][r] * st.x + st.y, 0.f));
        pooled[((size_t)b * 8192 + q) * 256 + c] = f2bf(m);
      }
    }
  }
}

// ------------- MLP2: concat(pooled, feature1) -> 384 -> 256, transposed out -
__global__ __launch_bounds__(256) void mlp2_kernel(
    const ushort_t* __restrict__ pooled, const ushort_t* __restrict__ f1T,
    const float* __restrict__ w, const float* __restrict__ bn,
    float* __restrict__ out) {
  extern __shared__ char smem[];
  ushort_t* wB = (ushort_t*)smem;           // 256 x 384
  float2* sbn = (float2*)(wB + 256 * 384);
  ushort_t* aT = (ushort_t*)(sbn + 256);    // 8 waves x 16 x 384

  int tid = threadIdx.x, lane = tid & 31, wave = tid >> 5;
  for (int i = tid; i < 256 * 384; i += 256) wB[i] = f2bf(w[i]);
  if (tid < 256) {
    float s = bn[tid] * rsqrtf(bn[768 + tid] + EPS_BN);
    sbn[tid] = make_float2(s, bn[256 + tid] - bn[512 + tid] * s);
  }
  __syncthreads();

  ushort_t* aw = aT + wave * 16 * 384;
  int r0 = blockIdx.x * 128 + wave * 16;    // flat row = b*8192 + n
  for (int t = lane; t < 16 * 24; t += 32) {   // 24 chunks of 16 bf16 per row
    int row = t / 24, ch = (t - row * 24) * 16;
    size_t r = (size_t)(r0 + row);
    const ushort_t* src = (ch < 256) ? pooled + r * 256 + ch
                                     : f1T + r * 128 + (ch - 256);
    ushort_t* dst = aw + row * 384 + ch;
    async_copy_b128(src, dst);
    async_copy_b128(src + 8, dst + 8);
  }
  wait_async_lds();

  int b = r0 >> 13, nbase = r0 & 8191;
  int m0 = (lane >> 4) * 8;
#pragma unroll
  for (int h = 0; h < 2; ++h) {
    v8f acc[8];
#pragma unroll
    for (int nt = 0; nt < 8; ++nt)
      acc[nt] = gemm_tile<12>(aw, 384,
                              wB + ((h * 8 + nt) * 16 + (lane & 15)) * 384, lane);
#pragma unroll
    for (int nt = 0; nt < 8; ++nt) {
      int c = (h * 8 + nt) * 16 + (lane & 15);
      float2 st = sbn[c];
      float y[8];
#pragma unroll
      for (int r = 0; r < 8; ++r) y[r] = fmaxf(acc[nt][r] * st.x + st.y, 0.f);
      // lane's 8 accs are 8 consecutive n for one channel -> two float4 stores
      float4* po = (float4*)(out + ((size_t)b * 256 + c) * 8192 + nbase + m0);
      po[0] = make_float4(y[0], y[1], y[2], y[3]);
      po[1] = make_float4(y[4], y[5], y[6], y[7]);
    }
  }
}

extern "C" void kernel_launch(void* const* d_in, const int* in_sizes, int n_in,
                              void* d_out, int out_size, void* d_ws, size_t ws_size,
                              hipStream_t stream) {
  (void)in_sizes; (void)n_in; (void)out_size; (void)ws_size;
  const float* xyz1     = (const float*)d_in[0];
  const float* xyz2     = (const float*)d_in[1];
  const float* feature1 = (const float*)d_in[2];
  const float* feature2 = (const float*)d_in[3];
  const float* w1_0 = (const float*)d_in[4];
  const float* bn1_0 = (const float*)d_in[5];
  const float* w1_1 = (const float*)d_in[6];
  const float* bn1_1 = (const float*)d_in[7];
  const float* w1_2 = (const float*)d_in[8];
  const float* bn1_2 = (const float*)d_in[9];
  const float* w2_0 = (const float*)d_in[10];
  const float* bn2_0 = (const float*)d_in[11];
  float* out = (float*)d_out;

  char* ws = (char*)d_ws;
  int* idx         = (int*)ws;                                 // 1 MiB
  ushort_t* f2T    = (ushort_t*)(ws + (size_t)(1u << 20));     // 4 MiB
  ushort_t* f1T    = (ushort_t*)(ws + (size_t)(5u << 20));     // 8 MiB
  ushort_t* pooled = (ushort_t*)(ws + (size_t)(13u << 20));    // 16 MiB

  knn_kernel<<<dim3(32, 4), 256, 0, stream>>>(xyz1, xyz2, idx);
  transpose_cast_kernel<<<dim3(8, 256, 4), 256, 0, stream>>>(feature2, f2T, 256, 2048);
  transpose_cast_kernel<<<dim3(32, 128, 4), 256, 0, stream>>>(feature1, f1T, 128, 8192);

  size_t lds1 = (size_t)(128 * 288 + 128 * 128 + 256 * 128) * 2  // weights bf16
              + (size_t)(128 + 128 + 256) * 8                    // folded BN
              + (size_t)8 * 16 * 288 * 2;                        // A tiles
  mlp1_kernel<<<dim3(256, 4), 256, lds1, stream>>>(
      xyz1, xyz2, idx, f2T, w1_0, w1_1, w1_2, bn1_0, bn1_1, bn1_2, pooled);

  size_t lds2 = (size_t)256 * 384 * 2 + 256 * 8 + (size_t)8 * 16 * 384 * 2;
  mlp2_kernel<<<256, 256, lds2, stream>>>(pooled, f1T, w2_0, bn2_0, out);
}
